// VectorQuantizerEMA_10763188044255
// MI455X (gfx1250) — matrix operand order repaired and checked by reading, not compiled, not run on hardware
//
#include <hip/hip_runtime.h>
#include <hip/hip_bf16.h>
#include <math.h>

typedef __attribute__((ext_vector_type(16))) __bf16 v16bf;
typedef __attribute__((ext_vector_type(8)))  float  v8f;

#define NUM_EMB   512
#define EMB_DIM   64
#define N_ROWS    131072          // 32*64*64
#define ROWS_PER_BLK 64
#define NBLK      (N_ROWS / ROWS_PER_BLK)   // 2048
#define TOTAL_ELEMS 8388608.0f    // 32*64*64*64

struct __align__(16) US8 { unsigned short u[8]; };

// ---------------- prep: bf16 hi/lo split of embedding + |e|^2 + zero counts --------
__global__ void vq_prep(const float* __restrict__ emb,
                        unsigned short* __restrict__ emb_hi,
                        unsigned short* __restrict__ emb_lo,
                        float* __restrict__ enorm,
                        unsigned* __restrict__ counts) {
  int t = blockIdx.x * blockDim.x + threadIdx.x;
  if (t < NUM_EMB) {
    const float* row = emb + t * EMB_DIM;
    float s = 0.f;
    #pragma unroll 8
    for (int d = 0; d < EMB_DIM; ++d) {
      float f = row[d];
      s += f * f;
      __bf16 h = (__bf16)f;
      float lo = f - (float)h;
      emb_hi[t * EMB_DIM + d] = __builtin_bit_cast(unsigned short, h);
      emb_lo[t * EMB_DIM + d] = __builtin_bit_cast(unsigned short, (__bf16)lo);
    }
    enorm[t]  = s;
    counts[t] = 0u;
  }
}

// ---------------- main: distances via bf16x3 WMMA, argmin, gather, outputs ---------
__global__ __launch_bounds__(128) void vq_main(
    const float* __restrict__ x_nchw,           // [32][64][64][64]  (B,C,H,W)
    const float* __restrict__ emb,              // [512][64] f32
    const unsigned short* __restrict__ emb_hi,  // bf16 bits [512][64]
    const unsigned short* __restrict__ emb_lo,  // bf16 bits [512][64]
    const float* __restrict__ enorm_g,          // [512]
    unsigned* __restrict__ counts,              // [512]
    float* __restrict__ partials,               // [NBLK] per-block loss partial
    float* __restrict__ out_nchw,               // [32][64][64][64]
    float* __restrict__ out_nhwc) {             // [131072][64]

  __shared__ float    xs[ROWS_PER_BLK][EMB_DIM + 1]; // padded vs bank conflicts
  __shared__ float    enorm_s[NUM_EMB];
  __shared__ unsigned short idx_s[ROWS_PER_BLK];
  __shared__ unsigned hist[NUM_EMB];
  __shared__ float    red[128];

  const int t    = threadIdx.x;
  const int lane = t & 31;
  const int wq   = t >> 5;                 // wave 0..3
  const int bh   = blockIdx.x;             // block == one (b,h) W-line of 64 rows
  const int b    = bh >> 6;
  const int h    = bh & 63;
  const int n0   = bh * ROWS_PER_BLK;      // first flat row (n = (b*64+h)*64 + w)

  // zero histogram, cache code norms
  for (int i = t; i < NUM_EMB; i += 128) { hist[i] = 0u; enorm_s[i] = enorm_g[i]; }

  // stage X tile (transpose NCHW -> xs[w][c]); lanes sweep w => coalesced
  const float* inb = x_nchw + (size_t)b * (EMB_DIM * 4096) + (size_t)h * 64;
  for (int v = t; v < ROWS_PER_BLK * EMB_DIM; v += 128) {
    int c = v >> 6, w = v & 63;
    xs[w][c] = inb[(size_t)c * 4096 + w];
  }
  __syncthreads();

  // ---- B fragments: this wave's 16 rows, bf16 hi/lo, K chunks j=0,1 ----
  const int nrow = lane & 15;
  const int row  = wq * 16 + nrow;
  v16bf Bh[2], Bl[2];
  #pragma unroll
  for (int j = 0; j < 2; ++j) {
    int d0 = j * 32 + ((lane >> 4) << 3);  // lanes 0-15: K 0..7,16..23, lanes 16-31: K 8..15,24..31
    #pragma unroll
    for (int i = 0; i < 16; ++i) {
      float f = xs[row][i < 8 ? d0 + i : d0 + 8 + i];
      __bf16 hb = (__bf16)f;
      Bh[j][i] = hb;
      Bl[j][i] = (__bf16)(f - (float)hb);
    }
  }

  // ---- sweep 32 code tiles; A = embedding tile; dist = |e|^2 - 2 e.x ----
  // Double-buffered A fragments: issue tile kt+1 loads before consuming tile kt,
  // so the per-iteration wait is loadcnt<=8 (overlap L2 latency with WMMA+argmin).
  const int half8 = (lane >> 4) << 3;      // 0 for lanes 0-15, 8 for 16-31

  US8 frag[2][8];
  auto loadfrag = [&](int buf, int kt) {
    #pragma unroll
    for (int j = 0; j < 2; ++j) {
      int abase = (kt * 16 + nrow) * EMB_DIM + j * 32 + half8;
      frag[buf][j * 4 + 0] = *(const US8*)(emb_hi + abase);
      frag[buf][j * 4 + 1] = *(const US8*)(emb_hi + abase + 16);
      frag[buf][j * 4 + 2] = *(const US8*)(emb_lo + abase);
      frag[buf][j * 4 + 3] = *(const US8*)(emb_lo + abase + 16);
    }
  };
  auto mkv = [](const US8& a, const US8& b) {
    v16bf v;
    #pragma unroll
    for (int i = 0; i < 8; ++i) {
      v[i]     = __builtin_bit_cast(__bf16, a.u[i]);
      v[i + 8] = __builtin_bit_cast(__bf16, b.u[i]);
    }
    return v;
  };

  float bestd = 3.4e38f;
  int   besti = 0;
  loadfrag(0, 0);
  #pragma unroll 2
  for (int kt = 0; kt < 32; ++kt) {
    const int pb = kt & 1;                 // static after unroll-by-2
    if (kt < 31) loadfrag(pb ^ 1, kt + 1); // prefetch next tile's fragments
    v8f C = {};
    #pragma unroll
    for (int j = 0; j < 2; ++j) {
      v16bf Ah = mkv(frag[pb][j * 4 + 0], frag[pb][j * 4 + 1]);
      v16bf Al = mkv(frag[pb][j * 4 + 2], frag[pb][j * 4 + 3]);
      C = __builtin_amdgcn_wmma_f32_16x16x32_bf16(false, Ah, false, Bh[j], (short)0, C, false, false);
      C = __builtin_amdgcn_wmma_f32_16x16x32_bf16(false, Ah, false, Bl[j], (short)0, C, false, false);
      C = __builtin_amdgcn_wmma_f32_16x16x32_bf16(false, Al, false, Bh[j], (short)0, C, false, false);
    }
    int mbase = kt * 16 + half8;           // codes this lane-half holds
    #pragma unroll
    for (int jj = 0; jj < 8; ++jj) {
      float dist = enorm_s[mbase + jj] - 2.0f * C[jj];
      if (dist < bestd) { bestd = dist; besti = mbase + jj; }  // ascending scan => first-min
    }
  }
  // merge the two lane-halves (codes =0..7 vs 8..15 mod 16), tie -> lower index
  float od = __shfl_xor(bestd, 16, 32);
  int   oi = __shfl_xor(besti, 16, 32);
  if (od < bestd || (od == bestd && oi < besti)) { bestd = od; besti = oi; }
  if (lane < 16) idx_s[row] = (unsigned short)besti;
  __syncthreads();

  // ---- pass 1: gather q, loss, STE value; write NHWC; stash val back in xs ----
  float lsum = 0.f;
  for (int v = t; v < ROWS_PER_BLK * EMB_DIM; v += 128) {
    int r = v >> 6, c = v & 63;
    float q  = emb[(int)idx_s[r] * EMB_DIM + c];
    float xv = xs[r][c];
    float dq = q - xv;
    lsum += dq * dq;
    float val = xv + dq;                               // straight-through forward
    out_nhwc[(size_t)(n0 + r) * EMB_DIM + c] = val;
    xs[r][c] = val;
  }
  if (t < ROWS_PER_BLK) atomicAdd(&hist[idx_s[t]], 1u);
  __syncthreads();

  // ---- pass 2: NCHW store (lanes sweep w => coalesced) ----
  float* ob = out_nchw + (size_t)b * (EMB_DIM * 4096) + (size_t)h * 64;
  for (int v = t; v < ROWS_PER_BLK * EMB_DIM; v += 128) {
    int c = v >> 6, w = v & 63;
    ob[(size_t)c * 4096 + w] = xs[w][c];
  }
  for (int i = t; i < NUM_EMB; i += 128) {
    unsigned n = hist[i];
    if (n) atomicAdd(&counts[i], n);
  }

  // deterministic per-block loss partial
  red[t] = lsum;
  __syncthreads();
  for (int s = 64; s > 0; s >>= 1) { if (t < s) red[t] += red[t + s]; __syncthreads(); }
  if (t == 0) partials[blockIdx.x] = red[0];
}

// ---------------- finalize: loss + perplexity (fixed-order reductions) ------------
__global__ void vq_finalize(const float* __restrict__ partials, int nparts,
                            const unsigned* __restrict__ counts,
                            float* __restrict__ out_loss, float* __restrict__ out_perp) {
  __shared__ float red[256];
  int t = threadIdx.x;
  float s = 0.f;
  for (int i = t; i < nparts; i += 256) s += partials[i];
  red[t] = s;
  __syncthreads();
  for (int k = 128; k > 0; k >>= 1) { if (t < k) red[t] += red[t + k]; __syncthreads(); }
  float total = red[0];

  float e = 0.f;
  for (int i = t; i < NUM_EMB; i += 256) {
    float p = (float)counts[i] / (float)N_ROWS;
    e += p * logf(p + 1e-10f);
  }
  __syncthreads();
  red[t] = e;
  __syncthreads();
  for (int k = 128; k > 0; k >>= 1) { if (t < k) red[t] += red[t + k]; __syncthreads(); }
  if (t == 0) {
    *out_loss = 0.25f * (total / TOTAL_ELEMS);
    *out_perp = expf(-red[0]);
  }
}

extern "C" void kernel_launch(void* const* d_in, const int* in_sizes, int n_in,
                              void* d_out, int out_size, void* d_ws, size_t ws_size,
                              hipStream_t stream) {
  (void)in_sizes; (void)n_in; (void)out_size; (void)ws_size;
  const float* x   = (const float*)d_in[0];   // [32,64,64,64] NCHW
  const float* emb = (const float*)d_in[1];   // [512,64]

  char* ws = (char*)d_ws;
  unsigned short* emb_hi = (unsigned short*)(ws);            // 64 KB
  unsigned short* emb_lo = (unsigned short*)(ws + 65536);    // 64 KB
  float*    enorm   = (float*)   (ws + 131072);              // 2 KB
  unsigned* counts  = (unsigned*)(ws + 133120);              // 2 KB
  float*    partials= (float*)   (ws + 135168);              // 8 KB

  float* out      = (float*)d_out;
  float* out_loss = out;                      // [0]
  float* out_nchw = out + 1;                  // [1 .. 1+8388608)
  float* out_perp = out + 1 + 8388608;        // [8388609]
  float* out_nhwc = out + 2 + 8388608;        // [8388610 ..)

  vq_prep    <<<2, 256, 0, stream>>>(emb, emb_hi, emb_lo, enorm, counts);
  vq_main    <<<NBLK, 128, 0, stream>>>(x, emb, emb_hi, emb_lo, enorm,
                                        counts, partials, out_nchw, out_nhwc);
  vq_finalize<<<1, 256, 0, stream>>>(partials, NBLK, counts, out_loss, out_perp);
}
